// MultiHeadAttention_39728447488403
// MI455X (gfx1250) — matrix operand order repaired and checked by reading, not compiled, not run on hardware
//
#include <hip/hip_runtime.h>
#include <hip/hip_bf16.h>

#define HH  16
#define NN  64
#define HIDD 1024
#define BB  2
#define LL  2048   // LF == LT

typedef __attribute__((ext_vector_type(16))) __bf16 v16bf;
typedef __attribute__((ext_vector_type(8)))  float  v8f;

union Frag16 {            // 16 bf16 operand fragment = 8 VGPRs
    v16bf v;
    uint4 u[2];
};
union FragC {             // 16x16 f32 C/D fragment slice = 8 VGPRs
    v8f   v;
    float f[8];
};

__device__ __forceinline__ unsigned short f2bf(float f) {
    unsigned int x = __float_as_uint(f);
    x += 0x7FFFu + ((x >> 16) & 1u);      // round-to-nearest-even
    return (unsigned short)(x >> 16);
}

// ---------------------------------------------------------------- convert f32 -> bf16
__global__ void cvt_bf16_kernel(const float* __restrict__ src,
                                unsigned short* __restrict__ dst, int n4) {
    int i = blockIdx.x * blockDim.x + threadIdx.x;
    if (i < n4) {
        float4 f = ((const float4*)src)[i];
        ushort4 o;
        o.x = f2bf(f.x); o.y = f2bf(f.y); o.z = f2bf(f.z); o.w = f2bf(f.w);
        ((ushort4*)dst)[i] = o;
    }
}

// ---------------------------------------------------- weight transpose: WT[n][k] = bf16(W[k][n])
__global__ void wtrans_kernel(const float* __restrict__ W,
                              unsigned short* __restrict__ WT) {
    __shared__ float tile[32][33];
    int n0 = blockIdx.x * 32, k0 = blockIdx.y * 32;
    int tx = threadIdx.x & 31, ty = threadIdx.x >> 5;     // 256 threads
    #pragma unroll
    for (int r = ty; r < 32; r += 8)
        tile[r][tx] = W[(size_t)(k0 + r) * HIDD + (n0 + tx)];
    __syncthreads();
    #pragma unroll
    for (int r = ty; r < 32; r += 8)
        WT[(size_t)(n0 + r) * HIDD + (k0 + tx)] = f2bf(tile[tx][r]);
}

// ---------------------------------------------------------------- GEMM: Y = X @ W + b
// X: [M, HIDD] bf16, WT: [HIDD, HIDD] bf16 (WT[n][k] = W[k][n]).
// mode 0: out = bf16 heads [B,H,L,64]   (Q, K)
// mode 1: out = bf16 heads transposed [B,H,64,L]   (V)
// mode 2: out = f32 [M, HIDD]          (final projection)
__global__ __launch_bounds__(128)
void gemm_kernel(const unsigned short* __restrict__ X,
                 const unsigned short* __restrict__ WT,
                 const float* __restrict__ bias,
                 void* __restrict__ out, int M, int mode) {
    int wave = threadIdx.x >> 5;
    int lane = threadIdx.x & 31;
    int l16  = lane & 15, g = lane >> 4;

    int tile   = blockIdx.x * 4 + wave;
    int ntilen = HIDD / 64;                      // 16
    int tm = (tile / ntilen) * 64;
    int tn = (tile % ntilen) * 64;
    if (tm >= M) return;

    FragC c[4][4];
    #pragma unroll
    for (int i = 0; i < 4; i++)
        #pragma unroll
        for (int j = 0; j < 4; j++)
            #pragma unroll
            for (int r = 0; r < 8; r++) c[i][j].f[r] = 0.f;

    const unsigned short* arow[4];
    const unsigned short* brow[4];
    #pragma unroll
    for (int i = 0; i < 4; i++)
        arow[i] = X  + (size_t)(tm + i * 16 + l16) * HIDD;
    #pragma unroll
    for (int j = 0; j < 4; j++)
        brow[j] = WT + (size_t)(tn + j * 16 + l16) * HIDD;

    for (int k0 = 0; k0 < HIDD; k0 += 32) {
        Frag16 a[4], b[4];
        #pragma unroll
        for (int i = 0; i < 4; i++) {            // A: lane-group K runs {8g..}, {16+8g..}
            a[i].u[0] = *(const uint4*)(arow[i] + k0 + 8 * g);
            a[i].u[1] = *(const uint4*)(arow[i] + k0 + 16 + 8 * g);
        }
        #pragma unroll
        for (int j = 0; j < 4; j++) {            // B: lane-group K runs {8g..}, {16+8g..}
            b[j].u[0] = *(const uint4*)(brow[j] + k0 + 8 * g);
            b[j].u[1] = *(const uint4*)(brow[j] + k0 + 16 + 8 * g);
        }
        #pragma unroll
        for (int i = 0; i < 4; i++)
            #pragma unroll
            for (int j = 0; j < 4; j++)
                c[i][j].v = __builtin_amdgcn_wmma_f32_16x16x32_bf16(
                    false, a[i].v, false, b[j].v, (short)0, c[i][j].v, false, false);
    }

    #pragma unroll
    for (int j = 0; j < 4; j++) {
        int nf = tn + j * 16 + l16;              // full output column
        float bj = bias[nf];
        int h = nf >> 6, n = nf & 63;
        #pragma unroll
        for (int i = 0; i < 4; i++) {
            #pragma unroll
            for (int r = 0; r < 8; r++) {
                int m = tm + i * 16 + r + 8 * g;
                float val = c[i][j].f[r] + bj;
                if (mode == 0) {
                    int bi = m / LL, l = m % LL;
                    ((unsigned short*)out)[(((size_t)(bi * HH + h)) * LL + l) * NN + n] = f2bf(val);
                } else if (mode == 1) {
                    int bi = m / LL, l = m % LL;
                    ((unsigned short*)out)[(((size_t)(bi * HH + h)) * NN + n) * LL + l] = f2bf(val);
                } else {
                    ((float*)out)[(size_t)m * HIDD + nf] = val;
                }
            }
        }
    }
}

// ---------------------------------------------------------------- attention (flash-style)
// One wave owns a 32-row f-strip (2 m-subtiles); K/V fragments loaded once per
// chunk and reused across both m-subtiles (32 FLOP/byte from L2).
// Qh, Kh: [B,H,L,64] bf16.  VT: [B,H,64,L] bf16.  Obuf: [B,L,HIDD] bf16.
__global__ __launch_bounds__(128)
void attn_kernel(const unsigned short* __restrict__ Qh,
                 const unsigned short* __restrict__ Kh,
                 const unsigned short* __restrict__ VT,
                 const int* __restrict__ masks,
                 unsigned short* __restrict__ Obuf) {
    __shared__ unsigned short Plds[4][32 * 32];  // 2KB private P tile per wave

    int wave = threadIdx.x >> 5, lane = threadIdx.x & 31;
    int l16 = lane & 15, g = lane >> 4;

    const int strips = LL / 32;                  // 64 f-strips per (b,h)
    int gid = blockIdx.x * 4 + wave;
    int bh  = gid / strips;
    int f0  = (gid % strips) * 32;
    int b   = bh / HH, h = bh % HH;

    const unsigned short* Qp = Qh + (size_t)bh * LL * NN;
    const unsigned short* Kp = Kh + (size_t)bh * LL * NN;
    const unsigned short* Vp = VT + (size_t)bh * NN * LL;
    const int* mp = masks + b * LL;

    // persistent Q A-fragments: 2 m-subtiles x 2 K-slabs over n=64
    Frag16 qf[2][2];
    #pragma unroll
    for (int i = 0; i < 2; i++) {
        const unsigned short* qrow = Qp + (size_t)(f0 + i * 16 + l16) * NN;
        #pragma unroll
        for (int s = 0; s < 2; s++) {
            qf[i][s].u[0] = *(const uint4*)(qrow + s * 32 + 8 * g);
            qf[i][s].u[1] = *(const uint4*)(qrow + s * 32 + 16 + 8 * g);
        }
    }

    float mrow[2][8], lrow[2][8];
    FragC o[2][4];
    #pragma unroll
    for (int i = 0; i < 2; i++) {
        #pragma unroll
        for (int r = 0; r < 8; r++) { mrow[i][r] = -1e30f; lrow[i][r] = 0.f; }
        #pragma unroll
        for (int j = 0; j < 4; j++)
            #pragma unroll
            for (int r = 0; r < 8; r++) o[i][j].f[r] = 0.f;
    }

    unsigned short* pl = Plds[wave];

    for (int t0 = 0; t0 < LL; t0 += 32) {
        // -------- scores: S[i][tt] = Q_i . K_tt^T  (K frags shared across i)
        FragC s[2][2];
        #pragma unroll
        for (int i = 0; i < 2; i++)
            #pragma unroll
            for (int tt = 0; tt < 2; tt++)
                #pragma unroll
                for (int r = 0; r < 8; r++) s[i][tt].f[r] = 0.f;

        #pragma unroll
        for (int tt = 0; tt < 2; tt++) {
            const unsigned short* krow = Kp + (size_t)(t0 + tt * 16 + l16) * NN;
            #pragma unroll
            for (int sl = 0; sl < 2; sl++) {
                Frag16 kf;
                kf.u[0] = *(const uint4*)(krow + sl * 32 + 8 * g);
                kf.u[1] = *(const uint4*)(krow + sl * 32 + 16 + 8 * g);
                #pragma unroll
                for (int i = 0; i < 2; i++)
                    s[i][tt].v = __builtin_amdgcn_wmma_f32_16x16x32_bf16(
                        false, qf[i][sl].v, false, kf.v, (short)0, s[i][tt].v, false, false);
            }
        }

        float bias0 = mp[t0 + l16]      ? 0.f : -1e12f;
        float bias1 = mp[t0 + 16 + l16] ? 0.f : -1e12f;

        // -------- online softmax per m-subtile (reduced across 16-lane group)
        #pragma unroll
        for (int i = 0; i < 2; i++) {
            float p0[8], p1[8], scl[8];
            #pragma unroll
            for (int r = 0; r < 8; r++) {
                float x0 = s[i][0].f[r] * 0.125f + bias0;    // 1/sqrt(64)
                float x1 = s[i][1].f[r] * 0.125f + bias1;
                float tmax = fmaxf(x0, x1);
                #pragma unroll
                for (int off = 1; off < 16; off <<= 1)
                    tmax = fmaxf(tmax, __shfl_xor(tmax, off, 16));
                float mn = fmaxf(mrow[i][r], tmax);
                float sc = __expf(mrow[i][r] - mn);
                float e0 = __expf(x0 - mn);
                float e1 = __expf(x1 - mn);
                float rs = e0 + e1;
                #pragma unroll
                for (int off = 1; off < 16; off <<= 1)
                    rs += __shfl_xor(rs, off, 16);
                lrow[i][r] = lrow[i][r] * sc + rs;
                mrow[i][r] = mn;
                scl[r] = sc;
                p0[r] = e0; p1[r] = e1;
            }
            #pragma unroll
            for (int j = 0; j < 4; j++)
                #pragma unroll
                for (int r = 0; r < 8; r++) o[i][j].f[r] *= scl[r];

            // C-frag -> A-frag relayout of P through private LDS slice
            #pragma unroll
            for (int r = 0; r < 8; r++) {
                pl[(i * 16 + r + 8 * g) * 32 + l16]      = f2bf(p0[r]);
                pl[(i * 16 + r + 8 * g) * 32 + 16 + l16] = f2bf(p1[r]);
            }
        }

        Frag16 pf[2];
        #pragma unroll
        for (int i = 0; i < 2; i++) {
            pf[i].u[0] = *(const uint4*)(pl + (i * 16 + l16) * 32 + 8 * g);
            pf[i].u[1] = *(const uint4*)(pl + (i * 16 + l16) * 32 + 16 + 8 * g);
        }

        // -------- O += P . V  (V frags shared across i)
        #pragma unroll
        for (int j = 0; j < 4; j++) {
            const unsigned short* vrow = Vp + (size_t)(j * 16 + l16) * LL + t0 + 8 * g;
            Frag16 vf;
            vf.u[0] = *(const uint4*)(vrow);
            vf.u[1] = *(const uint4*)(vrow + 16);
            #pragma unroll
            for (int i = 0; i < 2; i++)
                o[i][j].v = __builtin_amdgcn_wmma_f32_16x16x32_bf16(
                    false, pf[i].v, false, vf.v, (short)0, o[i][j].v, false, false);
        }
    }

    // -------- epilogue: O / l -> Obuf[b, f, h*64+n]
    #pragma unroll
    for (int i = 0; i < 2; i++)
        #pragma unroll
        for (int j = 0; j < 4; j++)
            #pragma unroll
            for (int r = 0; r < 8; r++) {
                int m = f0 + i * 16 + r + 8 * g;
                float val = o[i][j].f[r] / lrow[i][r];
                Obuf[((size_t)b * LL + m) * HIDD + h * NN + j * 16 + l16] = f2bf(val);
            }
}

// ---------------------------------------------------------------- host side
extern "C" void kernel_launch(void* const* d_in, const int* in_sizes, int n_in,
                              void* d_out, int out_size, void* d_ws, size_t ws_size,
                              hipStream_t stream) {
    (void)in_sizes; (void)n_in; (void)out_size; (void)ws_size;

    const float* q  = (const float*)d_in[0];
    const float* k  = (const float*)d_in[1];
    const float* v  = (const float*)d_in[2];
    const int*   mk = (const int*)  d_in[3];
    const float* Wq = (const float*)d_in[4];
    const float* bq = (const float*)d_in[5];
    const float* Wk = (const float*)d_in[6];
    const float* bk = (const float*)d_in[7];
    const float* Wv = (const float*)d_in[8];
    const float* bv = (const float*)d_in[9];
    const float* Wo = (const float*)d_in[10];
    const float* bo = (const float*)d_in[11];

    const size_t E = (size_t)BB * LL * HIDD;     // 4M elements per activation
    const size_t W = (size_t)HIDD * HIDD;        // 1M elements per weight

    unsigned short* p   = (unsigned short*)d_ws;
    unsigned short* qb  = p; p += E;             // bf16 activations
    unsigned short* kb  = p; p += E;
    unsigned short* vb  = p; p += E;
    unsigned short* WqT = p; p += W;             // bf16 transposed weights
    unsigned short* WkT = p; p += W;
    unsigned short* WvT = p; p += W;
    unsigned short* WoT = p; p += W;
    unsigned short* QhB = p; p += E;             // [B,H,L,64]
    unsigned short* KhB = p; p += E;             // [B,H,L,64]
    unsigned short* VTB = p; p += E;             // [B,H,64,L]
    unsigned short* Obf = qb;                    // reuse qb after projections

    const int M = BB * LL;                       // 4096 rows

    // 1) convert activations to bf16
    {
        int n4 = (int)(E / 4);
        int blocks = (n4 + 255) / 256;
        cvt_bf16_kernel<<<blocks, 256, 0, stream>>>(q, qb, n4);
        cvt_bf16_kernel<<<blocks, 256, 0, stream>>>(k, kb, n4);
        cvt_bf16_kernel<<<blocks, 256, 0, stream>>>(v, vb, n4);
    }
    // 2) transpose + convert weights
    {
        dim3 grid(HIDD / 32, HIDD / 32);
        wtrans_kernel<<<grid, 256, 0, stream>>>(Wq, WqT);
        wtrans_kernel<<<grid, 256, 0, stream>>>(Wk, WkT);
        wtrans_kernel<<<grid, 256, 0, stream>>>(Wv, WvT);
        wtrans_kernel<<<grid, 256, 0, stream>>>(Wo, WoT);
    }
    // 3) projections (64x64 tile per wave, 4 waves/block)
    {
        int tiles = (M / 64) * (HIDD / 64);      // 1024
        int blocks = tiles / 4;                  // 256
        gemm_kernel<<<blocks, 128, 0, stream>>>(qb, WqT, bq, (void*)QhB, M, 0);
        gemm_kernel<<<blocks, 128, 0, stream>>>(kb, WkT, bk, (void*)KhB, M, 0);
        gemm_kernel<<<blocks, 128, 0, stream>>>(vb, WvT, bv, (void*)VTB, M, 1);
    }
    // 4) attention: one wave per 32-row f-strip
    {
        int waves = BB * HH * (LL / 32);         // 2048
        attn_kernel<<<waves / 4, 128, 0, stream>>>(QhB, KhB, VTB, mk, Obf);
    }
    // 5) output projection -> f32 d_out
    {
        int tiles = (M / 64) * (HIDD / 64);
        gemm_kernel<<<tiles / 4, 128, 0, stream>>>(Obf, WoT, bo, d_out, M, 2);
    }
}
